// VariableSelectionNetwork_36369783062948
// MI455X (gfx1250) — compile-verified, matrix operand order checked by reading, not run on hardware
//
#include <hip/hip_runtime.h>
#include <hip/hip_bf16.h>
#include <math.h>

// ---------------------------------------------------------------------------
// VariableSelectionNetwork fused pipeline for gfx1250 (wave32, WMMA, async-LDS)
// B=32 S=256 F=64 U=128  -> N = 8192 rows, K = F*U = 8192, 3 GEMMs to 128 cols
// ---------------------------------------------------------------------------

typedef __attribute__((ext_vector_type(16))) __bf16 v16bf;
typedef __attribute__((ext_vector_type(8)))  float  v8f;

#define NROWS   8192      // B*S
#define KDIM    8192      // F*U
#define NKB     256       // KDIM / 32
#define NTILES  24        // 3 matrices * (128/16) col tiles
#define FRAG    512       // 32 lanes * 16 halves per fragment

// A/B 16-bit fragment K index for (half h, lane-hi) per CDNA5 ISA 7.12.2
__device__ __forceinline__ int kmap(int h, int hi) {
  return (h < 8) ? (hi * 8 + h) : (16 + hi * 8 + (h - 8));
}

// --- CDNA5 async global->LDS copy (ASYNCcnt), see CDNA5 ISA
__device__ __forceinline__ void async_copy_b128(unsigned lds_off, const void* gptr) {
  asm volatile("global_load_async_to_lds_b128 %0, %1, off"
               :: "v"(lds_off), "v"(gptr) : "memory");
}
__device__ __forceinline__ void wait_async0() {
  asm volatile("s_wait_asynccnt 0x0" ::: "memory");
}

// ---------------------------------------------------------------------------
// Pack elu_w / gate_w / proj_w ([8192,128] fp32) into bf16 WMMA B fragments:
// layout [kb][g=0..23][lane][h], one thread per half-pair.
// ---------------------------------------------------------------------------
__global__ __launch_bounds__(256) void vsn_pack_w(
    const float* __restrict__ ew, const float* __restrict__ gw,
    const float* __restrict__ pw, __bf16* __restrict__ wp) {
  size_t t = (size_t)blockIdx.x * 256 + threadIdx.x;   // 1,572,864 total
  int hp   = (int)(t & 7);  size_t r = t >> 3;
  int lane = (int)(r & 31); r >>= 5;
  int g    = (int)(r % 24);
  int kb   = (int)(r / 24);
  if (kb >= NKB) return;
  const float* W = (g < 8) ? ew : ((g < 16) ? gw : pw);
  int n  = (g & 7) * 16 + (lane & 15);
  int hi = lane >> 4;
#pragma unroll
  for (int b = 0; b < 2; ++b) {
    int h = hp * 2 + b;
    int k = kb * 32 + kmap(h, hi);
    wp[t * 2 + b] = (__bf16)W[(size_t)k * 128 + n];
  }
}

// ---------------------------------------------------------------------------
// Pack fw/fb ([64,128] fp32) into fp32 A-fragment lane layout [kb][lane][h]
// (pre-GELU, so must stay fp32).
// ---------------------------------------------------------------------------
__global__ __launch_bounds__(256) void vsn_pack_f(
    const float* __restrict__ fw, const float* __restrict__ fb,
    float* __restrict__ fwp, float* __restrict__ fbp) {
  int t = blockIdx.x * 256 + threadIdx.x;              // 131,072 total
  int h = t & 15, lane = (t >> 4) & 31, kb = t >> 9;
  if (kb >= NKB) return;
  int k = kb * 32 + kmap(h, lane >> 4);
  int f = k >> 7, u = k & 127;
  fwp[t] = fw[f * 128 + u];
  fbp[t] = fb[f * 128 + u];
}

// ---------------------------------------------------------------------------
// Main fused GEMM: rows of concat (built on the fly: gelu(x*fw+fb)) times the
// three stacked weight matrices. Workgroup = 64 rows x 384 cols, 8 waves.
// Double-buffered LDS tiles filled by async global->LDS DMA.
// ---------------------------------------------------------------------------
__global__ __launch_bounds__(256, 1) void vsn_gemm(
    const float* __restrict__ x, const float* __restrict__ fwp,
    const float* __restrict__ fbp, const __bf16* __restrict__ wp,
    float* __restrict__ gout) {
  __shared__ alignas(32) __bf16 Bs[2][NTILES * FRAG];   // 2 x 24 KB
  __shared__ alignas(16) float  Fs[2][FRAG];            // 2 x 2 KB
  __shared__ alignas(16) float  Fbs[2][FRAG];           // 2 x 2 KB

  const int tid  = threadIdx.x;
  const int wave = tid >> 5, lane = tid & 31;
  const int ml = lane & 15, hi = lane >> 4;
  const int mt = wave >> 1, half = wave & 1;
  const int rowBase = blockIdx.x * 64 + mt * 16;
  const int row = rowBase + ml;

  // Per-thread staging addresses. LDS byte offset = low 32 bits of the
  // generic shared-pointer (flat-LDS addressing truncates to addr[31:0]).
  const unsigned ldsB[2] = {
      (unsigned)(uintptr_t)&Bs[0][0] + (unsigned)tid * 16u,
      (unsigned)(uintptr_t)&Bs[1][0] + (unsigned)tid * 16u };
  const unsigned ldsF[2] = {
      (tid < 128) ? (unsigned)(uintptr_t)&Fs[0][0]  + (unsigned)tid * 16u
                  : (unsigned)(uintptr_t)&Fbs[0][0] + (unsigned)(tid - 128) * 16u,
      (tid < 128) ? (unsigned)(uintptr_t)&Fs[1][0]  + (unsigned)tid * 16u
                  : (unsigned)(uintptr_t)&Fbs[1][0] + (unsigned)(tid - 128) * 16u };
  const char* gB = (const char*)wp + (size_t)tid * 16;            // + kb*24576
  const char* gF = (tid < 128) ? (const char*)fwp + (size_t)tid * 16
                               : (const char*)fbp + (size_t)(tid - 128) * 16;  // + kb*2048

  v8f acc[12] = {};

  // prologue: stage kb = 0 into buffer 0
  {
#pragma unroll
    for (int i = 0; i < 6; ++i)
      async_copy_b128(ldsB[0] + (unsigned)i * 4096u, gB + i * 4096);
    async_copy_b128(ldsF[0], gF);
    wait_async0();
    __syncthreads();
  }

  for (int kb = 0; kb < NKB; ++kb) {
    const int cur = kb & 1;
    // ---- issue async DMA for kb+1 into the other buffer (overlaps compute)
    if (kb + 1 < NKB) {
      const char* gBn = gB + (size_t)(kb + 1) * 24576;
#pragma unroll
      for (int i = 0; i < 6; ++i)
        async_copy_b128(ldsB[cur ^ 1] + (unsigned)i * 4096u, gBn + i * 4096);
      async_copy_b128(ldsF[cur ^ 1], gF + (size_t)(kb + 1) * 2048);
    }

    // ---- A fragment in registers: gelu(x[row,f] * fw + fb), exact GELU
    const float xs = x[(size_t)row * 64 + (kb >> 2)];
    const float* fs  = &Fs[cur][lane * 16];
    const float* fbs = &Fbs[cur][lane * 16];
    v16bf a;
#pragma unroll
    for (int h = 0; h < 16; ++h) {
      float vv = fmaf(xs, fs[h], fbs[h]);
      float ge = 0.5f * vv * (1.0f + erff(vv * 0.70710678118654752f));
      a[h] = (__bf16)ge;
    }

    // ---- 12 WMMAs; ping-pong B fragment regs so ds_load(j+1) overlaps wmma(j)
    const __bf16* bbase = &Bs[cur][half * 12 * FRAG + lane * 16];
    v16bf bcur = *(const v16bf*)bbase;
#pragma unroll
    for (int j = 0; j < 12; ++j) {
      v16bf bnext = bcur;
      if (j < 11) bnext = *(const v16bf*)(bbase + (j + 1) * FRAG);
      acc[j] = __builtin_amdgcn_wmma_f32_16x16x32_bf16(
          false, a, false, bcur, (short)0, acc[j], false, false);
      bcur = bnext;
    }

    wait_async0();
    __syncthreads();
  }

  // ---- store: C layout VGPR r -> M = hi*8+r, N = ml
#pragma unroll
  for (int j = 0; j < 12; ++j) {
    int gcol = (half * 12 + j) * 16 + ml;
    int w = gcol >> 7, c = gcol & 127;
#pragma unroll
    for (int rr = 0; rr < 8; ++rr) {
      int orow = rowBase + hi * 8 + rr;
      gout[((size_t)w * NROWS + orow) * 128 + c] = acc[j][rr];
    }
  }
}

// ---------------------------------------------------------------------------
// Per-row epilogue: ELU, 128x128 matvec, sigmoid gate, residual, LayerNorm,
// softmax over F=64, weighted output. 128 threads, 16 rows per block.
// ---------------------------------------------------------------------------
#define ROWS_PER_BLK 16

__global__ __launch_bounds__(128) void vsn_epilogue(
    const float* __restrict__ gout, const float* __restrict__ x,
    const float* __restrict__ elu_b, const float* __restrict__ lin_w,
    const float* __restrict__ lin_b, const float* __restrict__ gate_b,
    const float* __restrict__ ln_g, const float* __restrict__ ln_b,
    const float* __restrict__ sm_w, const float* __restrict__ sm_b,
    float* __restrict__ out) {
  __shared__ float lw[128 * 128];   // 64 KB (of 320 KB LDS)
  __shared__ float h1s[128];
  __shared__ float ys[128];
  __shared__ float red[128];

  const int tid = threadIdx.x;
  for (int i = tid; i < 128 * 128; i += 128) lw[i] = lin_w[i];
  const float eb = elu_b[tid], lb = lin_b[tid], gb = gate_b[tid];
  const float lng = ln_g[tid], lnb = ln_b[tid];
  __syncthreads();

  for (int r = 0; r < ROWS_PER_BLK; ++r) {
    const int row = blockIdx.x * ROWS_PER_BLK + r;
    const float* e = gout + (size_t)row * 128;
    const float* g = gout + ((size_t)NROWS + row) * 128;
    const float* p = gout + ((size_t)2 * NROWS + row) * 128;

    float ev = e[tid] + eb;
    float h1 = ev > 0.0f ? ev : (expf(ev) - 1.0f);   // ELU
    h1s[tid] = h1;
    __syncthreads();

    float acc = lb;
#pragma unroll 8
    for (int v = 0; v < 128; ++v) acc = fmaf(h1s[v], lw[v * 128 + tid], acc);

    float gate = 1.0f / (1.0f + expf(-(g[tid] + gb)));
    float hf = gate * acc + p[tid];

    // LayerNorm (eps = 1e-3, keras default)
    red[tid] = hf; __syncthreads();
    for (int s = 64; s > 0; s >>= 1) { if (tid < s) red[tid] += red[tid + s]; __syncthreads(); }
    float mu = red[0] * (1.0f / 128.0f); __syncthreads();
    float d = hf - mu;
    red[tid] = d * d; __syncthreads();
    for (int s = 64; s > 0; s >>= 1) { if (tid < s) red[tid] += red[tid + s]; __syncthreads(); }
    float var = red[0] * (1.0f / 128.0f); __syncthreads();
    float y = d * rsqrtf(var + 1e-3f) * lng + lnb;
    ys[tid] = y;
    __syncthreads();

    // softmax over F = 64
    float logit = 0.0f;
    if (tid < 64) {
      logit = sm_b[tid];
      for (int u = 0; u < 128; ++u) logit = fmaf(ys[u], sm_w[u * 64 + tid], logit);
      red[tid] = logit;
    }
    __syncthreads();
    for (int s = 32; s > 0; s >>= 1) { if (tid < s) red[tid] = fmaxf(red[tid], red[tid + s]); __syncthreads(); }
    float mx = red[0]; __syncthreads();
    float ex = 0.0f;
    if (tid < 64) { ex = expf(logit - mx); red[tid] = ex; }
    __syncthreads();
    for (int s = 32; s > 0; s >>= 1) { if (tid < s) red[tid] += red[tid + s]; __syncthreads(); }
    float sum = red[0]; __syncthreads();

    if (tid < 64) {
      float wgt = ex / sum;
      out[(size_t)row * 64 + tid] = x[(size_t)row * 64 + tid] * wgt;           // x * weights
      out[(size_t)NROWS * 64 + (size_t)row * 64 + tid] = wgt;                  // weights
    }
    __syncthreads();
  }
}

// ---------------------------------------------------------------------------
// Launch
// ---------------------------------------------------------------------------
extern "C" void kernel_launch(void* const* d_in, const int* in_sizes, int n_in,
                              void* d_out, int out_size, void* d_ws, size_t ws_size,
                              hipStream_t stream) {
  const float* x      = (const float*)d_in[0];
  const float* fw     = (const float*)d_in[1];
  const float* fb     = (const float*)d_in[2];
  const float* elu_w  = (const float*)d_in[3];
  const float* elu_b  = (const float*)d_in[4];
  const float* lin_w  = (const float*)d_in[5];
  const float* lin_b  = (const float*)d_in[6];
  const float* gate_w = (const float*)d_in[7];
  const float* gate_b = (const float*)d_in[8];
  const float* proj_w = (const float*)d_in[9];
  const float* ln_g   = (const float*)d_in[10];
  const float* ln_b   = (const float*)d_in[11];
  const float* sm_w   = (const float*)d_in[12];
  const float* sm_b   = (const float*)d_in[13];

  char* ws = (char*)d_ws;
  __bf16* wp  = (__bf16*)ws;                   //  6,291,456 B packed bf16 weights
  float*  fwp = (float*)(ws + 6291456);        //    524,288 B packed fw
  float*  fbp = (float*)(ws + 6815744);        //    524,288 B packed fb
  float*  gout= (float*)(ws + 7340032);        // 12,582,912 B gemm outputs [3][8192][128]

  vsn_pack_w<<<6144, 256, 0, stream>>>(elu_w, gate_w, proj_w, wp);
  vsn_pack_f<<<512, 256, 0, stream>>>(fw, fb, fwp, fbp);
  vsn_gemm<<<NROWS / 64, 256, 0, stream>>>(x, fwp, fbp, wp, gout);
  vsn_epilogue<<<NROWS / ROWS_PER_BLK, 128, 0, stream>>>(
      gout, x, elu_b, lin_w, lin_b, gate_b, ln_g, ln_b, sm_w, sm_b, (float*)d_out);
}
